// Attention_86947317940821
// MI455X (gfx1250) — compile-verified, matrix operand order checked by reading, not run on hardware
//
#include <hip/hip_runtime.h>
#include <hip/hip_bf16.h>

typedef _Float16 f16;
typedef __attribute__((ext_vector_type(16))) _Float16 v16h;
typedef __attribute__((ext_vector_type(8)))  float    v8f;
typedef __attribute__((ext_vector_type(4)))  unsigned int u32x4;
typedef __attribute__((ext_vector_type(8)))  int      i32x8;
typedef __attribute__((ext_vector_type(4)))  int      i32x4;

#if defined(__has_builtin)
#if __has_builtin(__builtin_amdgcn_tensor_load_to_lds) && \
    __has_builtin(__builtin_amdgcn_s_wait_tensorcnt)
#define USE_TDM 1
#endif
#endif
#ifndef USE_TDM
#define USE_TDM 0
#endif

// ---------------------------------------------------------------------------
// WMMA helper (CDNA5 wave32, V_WMMA_F32_16X16X32_F16)
// ---------------------------------------------------------------------------
__device__ __forceinline__ v8f wmma16(v16h a, v16h b, v8f c) {
  return __builtin_amdgcn_wmma_f32_16x16x32_f16(false, a, false, b, (short)0, c,
                                                false, false);
}

// Load a 16x32 f16 A-fragment (or B^T-fragment) from an LDS tile.
// Lanes 0-15: M=lane, K pairs; lanes 16-31: same M, upper K half of each group.
__device__ __forceinline__ v16h ld_frag(const f16* __restrict__ buf, int row0,
                                        int col0, int stride) {
  const int lane = threadIdx.x & 31;
  const int m    = row0 + (lane & 15);
  const int half = lane >> 4;
  const f16* base = buf + m * stride + col0;
  v16h f;
#pragma unroll
  for (int v = 0; v < 8; ++v) {
    const int kk = ((v >> 2) << 4) + (half << 3) + ((v & 3) << 1);
    f[2 * v]     = base[kk];
    f[2 * v + 1] = base[kk + 1];
  }
  return f;
}

// ---------------------------------------------------------------------------
// TDM: 2D tile load Global -> LDS via tensor descriptor (ISA 08, D# packing).
// data_size = 2 bytes. Optional LDS row padding for bank-conflict-free tiles.
// Issue from ONE wave per workgroup (TDM ignores EXEC; one DMA per call).
// ---------------------------------------------------------------------------
#if USE_TDM
__device__ __forceinline__ void tdm_load_2d_f16(
    unsigned int lds_byte_off,   // LDS destination byte address
    const void* gsrc,            // global tile start
    unsigned int dim0,           // tensor width (elements)
    unsigned int dim1,           // tensor height (rows)
    unsigned int stride0,        // row stride (elements)
    unsigned int tile0,          // tile width (elements)
    unsigned int tile1,          // tile height (rows)
    unsigned int pad_en,         // 1 = pad LDS rows
    unsigned int pad_interval,   // code: 0=2dw,1=4,2=8,3=16,4=32,...
    unsigned int pad_amount) {   // code: N -> (N+1) dwords of pad
  const unsigned long long ga = (unsigned long long)gsrc;
  u32x4 g0;
  g0[0] = 1u;                                            // count=1 user desc
  g0[1] = lds_byte_off;                                  // lds_addr
  g0[2] = (unsigned int)(ga & 0xffffffffu);              // global_addr[31:0]
  g0[3] = (unsigned int)((ga >> 32) & 0x01ffffffu)       // global_addr[56:32]
          | (2u << 30);                                  // type=2 (image)
  i32x8 g1;
  g1[0] = (int)((1u << 16)                               // data_size=2B
                | (pad_en << 20)
                | (pad_interval << 22)
                | (pad_amount << 25));
  g1[1] = (int)(dim0 << 16);                             // tensor_dim0[15:0]
  g1[2] = (int)((dim0 >> 16) | (dim1 << 16));            // dim0 hi | dim1 lo
  g1[3] = (int)((dim1 >> 16) | (tile0 << 16));           // dim1 hi | tile_dim0
  g1[4] = (int)(tile1 & 0xffffu);                        // tile_dim1 (dim2=0)
  g1[5] = (int)stride0;                                  // dim0_stride[31:0]
  g1[6] = 0;
  g1[7] = 0;
  const i32x4 z4 = {0, 0, 0, 0};
#if __clang_major__ >= 23
  const i32x8 z8 = {0, 0, 0, 0, 0, 0, 0, 0};
  __builtin_amdgcn_tensor_load_to_lds(g0, g1, z4, z4, z8, 0);
#else
  __builtin_amdgcn_tensor_load_to_lds(g0, g1, z4, z4, 0);
#endif
}
#endif

// ---------------------------------------------------------------------------
// fp32 -> f16 conversion
// ---------------------------------------------------------------------------
__global__ __launch_bounds__(256) void cvt_f32_f16(const float* __restrict__ src,
                                                   f16* __restrict__ dst, int n) {
  const int i = blockIdx.x * blockDim.x + threadIdx.x;
  if (i < n) dst[i] = (f16)src[i];
}

// ---------------------------------------------------------------------------
// Tiled WMMA GEMM: C[MxN] = A[MxK]*B[KxN], f16 in, fp32 out.
// 256 threads (8 waves) -> 128x128 tile; wave -> 32x64 strip (8 WMMA/k-step).
// A tile staged by TDM (padded rows), B^T staged manually (transpose).
// Dynamic LDS: [0,12288) = A[128][48], [12288,24576) = B^T[128][48].
// ---------------------------------------------------------------------------
__global__ __launch_bounds__(256) void gemm_f16f32(const f16* __restrict__ A,
                                                   const f16* __restrict__ Bm,
                                                   float* __restrict__ C,
                                                   int N, int K) {
  constexpr int SA = 48;  // f16 row stride (96B, 16B aligned, +16 f16 pad)
  extern __shared__ __align__(16) char smem[];
  f16* ldsA  = (f16*)(smem);
  f16* ldsBT = (f16*)(smem + 128 * SA * sizeof(f16));

  const int tid   = threadIdx.x;
  const int wave  = tid >> 5;
  const int lane  = tid & 31;
  const int m0    = blockIdx.y * 128;
  const int n0    = blockIdx.x * 128;
  const int waveM = (wave & 3) * 32;
  const int waveN = (wave >> 2) * 64;

  v8f acc[2][4] = {};

  for (int k0 = 0; k0 < K; k0 += 32) {
#if USE_TDM
    if (wave == 0) {
      // 128-row x 32-elem tile, rows padded 16 dwords + 8 dwords -> stride 48
      tdm_load_2d_f16(0u, A + (size_t)m0 * K + k0,
                      /*dim0=*/32u, /*dim1=*/128u, /*stride0=*/(unsigned)K,
                      /*tile0=*/32u, /*tile1=*/128u,
                      /*pad_en=*/1u, /*pad_interval=*/3u, /*pad_amount=*/7u);
    }
#else
    {  // stage A 128x32: two float4 (16 f16) per thread
      const int row = tid >> 1;
      const int seg = (tid & 1) << 4;
      const f16* src = A + (size_t)(m0 + row) * K + k0 + seg;
      *(float4*)&ldsA[row * SA + seg]     = *(const float4*)src;
      *(float4*)&ldsA[row * SA + seg + 8] = *(const float4*)(src + 8);
    }
#endif
    {  // stage B^T: thread reads 16 contiguous n for one k, scatters to [n][k]
      const int kk   = tid >> 3;
      const int nseg = (tid & 7) << 4;
      const f16* src = Bm + (size_t)(k0 + kk) * N + n0 + nseg;
#pragma unroll
      for (int i = 0; i < 16; ++i) ldsBT[(nseg + i) * SA + kk] = src[i];
    }
#if USE_TDM
    if (wave == 0) __builtin_amdgcn_s_wait_tensorcnt(0);
#endif
    __syncthreads();

    const v16h a0 = ld_frag(ldsA, waveM, 0, SA);
    const v16h a1 = ld_frag(ldsA, waveM + 16, 0, SA);
    v16h b[4];
#pragma unroll
    for (int j = 0; j < 4; ++j) b[j] = ld_frag(ldsBT, waveN + 16 * j, 0, SA);
#pragma unroll
    for (int j = 0; j < 4; ++j) {
      acc[0][j] = wmma16(a0, b[j], acc[0][j]);
      acc[1][j] = wmma16(a1, b[j], acc[1][j]);
    }
    __syncthreads();
  }

  const int cn    = lane & 15;
  const int rbase = (lane >> 4) * 8;
#pragma unroll
  for (int i = 0; i < 2; ++i) {
#pragma unroll
    for (int r = 0; r < 8; ++r) {
      const size_t row = (size_t)(m0 + waveM + 16 * i + rbase + r);
#pragma unroll
      for (int j = 0; j < 4; ++j)
        C[row * N + n0 + waveN + 16 * j + cn] = acc[i][j][r];
    }
  }
}

// ---------------------------------------------------------------------------
// Rotary + q scale + head split: qkv fp32 [8192][1536] -> q,k,v f16 [B*H][N][64]
// ---------------------------------------------------------------------------
__global__ __launch_bounds__(256) void rotary_split(const float* __restrict__ qkv,
                                                    f16* __restrict__ q,
                                                    f16* __restrict__ k,
                                                    f16* __restrict__ v) {
  const int idx = blockIdx.x * blockDim.x + threadIdx.x;
  const int i = idx & 31;
  const int n = (idx >> 5) & 2047;
  const int h = (idx >> 16) & 7;
  const int b = idx >> 19;

  const size_t rowoff = (size_t)(b * 2048 + n) * 1536;
  const int col = h * 64 + 2 * i;

  const float q0 = qkv[rowoff + col],        q1 = qkv[rowoff + col + 1];
  const float k0 = qkv[rowoff + 512 + col],  k1 = qkv[rowoff + 512 + col + 1];
  const float v0 = qkv[rowoff + 1024 + col], v1 = qkv[rowoff + 1024 + col + 1];

  const float freq = (float)n * __expf(-(float)(2 * i) * (9.210340371976184f / 64.0f));
  float s, c;
  __sincosf(freq, &s, &c);
  const float scale = 0.125f;  // 64^-0.5

  const size_t o = ((size_t)(b * 8 + h) * 2048 + n) * 64 + 2 * i;
  q[o]     = (f16)((q0 * c - q1 * s) * scale);
  q[o + 1] = (f16)((q1 * c + q0 * s) * scale);
  k[o]     = (f16)(k0 * c - k1 * s);
  k[o + 1] = (f16)(k1 * c + k0 * s);
  v[o]     = (f16)v0;
  v[o + 1] = (f16)v1;
}

// ---------------------------------------------------------------------------
// Flash attention: block per (b*h, 64-row tile), 256 threads = 8 waves.
// Q/K tiles via TDM (padded to 72-halfword rows), V transposed manually,
// S = Q K^T + pos_bias via WMMA, online softmax in LDS, O += P V via WMMA.
// ---------------------------------------------------------------------------
#define FA_SD 72               // f16 row stride (144B)
#define FA_SS 68               // fp32 row stride for S
#define FA_TILE_B (64 * FA_SD * 2)
#define FA_OFF_Q  0
#define FA_OFF_K  (FA_OFF_Q + FA_TILE_B)
#define FA_OFF_VT (FA_OFF_K + FA_TILE_B)
#define FA_OFF_P  (FA_OFF_VT + FA_TILE_B)
#define FA_OFF_S  (FA_OFF_P + FA_TILE_B)
#define FA_OFF_M  (FA_OFF_S + 64 * FA_SS * 4)
#define FA_OFF_L  (FA_OFF_M + 64 * 4)
#define FA_OFF_A  (FA_OFF_L + 64 * 4)
#define FA_SMEM   (FA_OFF_A + 64 * 4)

__global__ __launch_bounds__(256) void flash_attn(const f16* __restrict__ q,
                                                  const f16* __restrict__ k,
                                                  const f16* __restrict__ v,
                                                  const float* __restrict__ pos_bias,
                                                  f16* __restrict__ out) {
  extern __shared__ __align__(16) char smem[];
  f16*   ldsQ  = (f16*)(smem + FA_OFF_Q);
  f16*   ldsK  = (f16*)(smem + FA_OFF_K);
  f16*   ldsVT = (f16*)(smem + FA_OFF_VT);
  f16*   ldsP  = (f16*)(smem + FA_OFF_P);
  float* ldsS  = (float*)(smem + FA_OFF_S);
  float* rowM  = (float*)(smem + FA_OFF_M);
  float* rowL  = (float*)(smem + FA_OFF_L);
  float* rowA  = (float*)(smem + FA_OFF_A);

  const int tid  = threadIdx.x;
  const int wave = tid >> 5;
  const int lane = tid & 31;
  const int bh   = blockIdx.y;  // b*8 + h
  const int h    = bh & 7;
  const int b    = bh >> 3;
  const int i0   = blockIdx.x * 64;
  const size_t headBase = (size_t)bh * 2048 * 64;

#if USE_TDM
  if (wave == 0) {
    // 64x64 f16 tile; rows padded 32 dwords + 4 dwords -> stride 72 halfwords
    tdm_load_2d_f16((unsigned)FA_OFF_Q, q + headBase + (size_t)i0 * 64,
                    64u, 2048u, 64u, 64u, 64u, 1u, 4u, 3u);
    __builtin_amdgcn_s_wait_tensorcnt(0);
  }
#else
  {  // stage Q tile 64x64 (16 f16 per thread)
    const int row = tid >> 2;
    const int seg = (tid & 3) << 4;
    const f16* src = q + headBase + (size_t)(i0 + row) * 64 + seg;
    *(float4*)&ldsQ[row * FA_SD + seg]     = *(const float4*)src;
    *(float4*)&ldsQ[row * FA_SD + seg + 8] = *(const float4*)(src + 8);
  }
#endif
  if (tid < 64) { rowM[tid] = -1e30f; rowL[tid] = 0.0f; }

  const int waveM = (wave & 3) * 16;
  const int waveN = (wave >> 2) * 32;
  v8f accO0 = {};
  v8f accO1 = {};
  __syncthreads();

  for (int j0 = 0; j0 < 2048; j0 += 64) {
#if USE_TDM
    if (wave == 0) {
      tdm_load_2d_f16((unsigned)FA_OFF_K, k + headBase + (size_t)j0 * 64,
                      64u, 2048u, 64u, 64u, 64u, 1u, 4u, 3u);
    }
#else
    {  // stage K row-major
      const int row = tid >> 2;
      const int seg = (tid & 3) << 4;
      const f16* ksrc = k + headBase + (size_t)(j0 + row) * 64 + seg;
      *(float4*)&ldsK[row * FA_SD + seg]     = *(const float4*)ksrc;
      *(float4*)&ldsK[row * FA_SD + seg + 8] = *(const float4*)(ksrc + 8);
    }
#endif
    {  // stage V transposed: [d][j]
      const int row = tid >> 2;
      const int seg = (tid & 3) << 4;
      if (j0 + 64 < 2048)
        __builtin_prefetch(v + headBase + (size_t)(j0 + 64 + row) * 64, 0, 1);
      const f16* vsrc = v + headBase + (size_t)(j0 + row) * 64 + seg;
#pragma unroll
      for (int t = 0; t < 16; ++t) ldsVT[(seg + t) * FA_SD + row] = vsrc[t];
    }
#if USE_TDM
    if (wave == 0) __builtin_amdgcn_s_wait_tensorcnt(0);
#endif
    __syncthreads();

    // S = Q K^T : each wave 2 tiles of 16x16, dim-64 in two 32-steps
    v8f s0 = {};
    v8f s1 = {};
#pragma unroll
    for (int d0 = 0; d0 < 64; d0 += 32) {
      const v16h aq  = ld_frag(ldsQ, waveM, d0, FA_SD);
      const v16h bk0 = ld_frag(ldsK, waveN, d0, FA_SD);
      const v16h bk1 = ld_frag(ldsK, waveN + 16, d0, FA_SD);
      s0 = wmma16(aq, bk0, s0);
      s1 = wmma16(aq, bk1, s1);
    }
    {  // + pos_bias (streamed once from HBM), spill S to LDS
      const int cn    = lane & 15;
      const int rbase = (lane >> 4) * 8;
      const float* pb = pos_bias + (size_t)h * 2048 * 2048;
#pragma unroll
      for (int r = 0; r < 8; ++r) {
        const int ri = waveM + rbase + r;
        const size_t prow = (size_t)(i0 + ri) * 2048 + j0;
        ldsS[ri * FA_SS + waveN + cn]      = s0[r] + pb[prow + waveN + cn];
        ldsS[ri * FA_SS + waveN + 16 + cn] = s1[r] + pb[prow + waveN + 16 + cn];
      }
    }
    __syncthreads();

    // online softmax: one row per thread (tid < 64)
    if (tid < 64) {
      const float mold = rowM[tid];
      float mj = mold;
      const float* srow = &ldsS[tid * FA_SS];
#pragma unroll
      for (int c = 0; c < 64; ++c) mj = fmaxf(mj, srow[c]);
      const float alpha = __expf(mold - mj);
      float lsum = 0.0f;
      f16* prow = &ldsP[tid * FA_SD];
#pragma unroll
      for (int c = 0; c < 64; ++c) {
        const float p = __expf(srow[c] - mj);
        lsum += p;
        prow[c] = (f16)p;
      }
      rowM[tid] = mj;
      rowL[tid] = rowL[tid] * alpha + lsum;
      rowA[tid] = alpha;
    }
    __syncthreads();

    // rescale running O, then O += P V
    {
      const int rbase = (lane >> 4) * 8;
#pragma unroll
      for (int r = 0; r < 8; ++r) {
        const float a = rowA[waveM + rbase + r];
        accO0[r] *= a;
        accO1[r] *= a;
      }
    }
#pragma unroll
    for (int c0 = 0; c0 < 64; c0 += 32) {
      const v16h ap  = ld_frag(ldsP, waveM, c0, FA_SD);
      const v16h bv0 = ld_frag(ldsVT, waveN, c0, FA_SD);
      const v16h bv1 = ld_frag(ldsVT, waveN + 16, c0, FA_SD);
      accO0 = wmma16(ap, bv0, accO0);
      accO1 = wmma16(ap, bv1, accO1);
    }
    __syncthreads();
  }

  {  // normalize and write out[b][i][h*64 + d] as f16
    const int cn    = lane & 15;
    const int rbase = (lane >> 4) * 8;
#pragma unroll
    for (int r = 0; r < 8; ++r) {
      const int ri = waveM + rbase + r;
      const float inv = 1.0f / rowL[ri];
      const size_t o = ((size_t)(b * 2048 + i0 + ri)) * 512 + h * 64;
      out[o + waveN + cn]      = (f16)(accO0[r] * inv);
      out[o + waveN + 16 + cn] = (f16)(accO1[r] * inv);
    }
  }
}

// ---------------------------------------------------------------------------
// Host launcher
// ---------------------------------------------------------------------------
extern "C" void kernel_launch(void* const* d_in, const int* in_sizes, int n_in,
                              void* d_out, int out_size, void* d_ws, size_t ws_size,
                              hipStream_t stream) {
  const float* x        = (const float*)d_in[0];  // [4][2048][512]
  const float* pos_bias = (const float*)d_in[1];  // [8][2048][2048]
  const float* W_qkv    = (const float*)d_in[2];  // [512][1536]
  const float* W_out    = (const float*)d_in[3];  // [512][512]
  float* out = (float*)d_out;                     // [4][2048][512]

  char* ws = (char*)d_ws;
  f16* xf     = (f16*)ws;  ws += (size_t)8192 * 512 * sizeof(f16);
  f16* wqkvf  = (f16*)ws;  ws += (size_t)512 * 1536 * sizeof(f16);
  f16* woutf  = (f16*)ws;  ws += (size_t)512 * 512 * sizeof(f16);
  float* qkv  = (float*)ws; ws += (size_t)8192 * 1536 * sizeof(float);
  f16* qf     = (f16*)ws;  ws += (size_t)32 * 2048 * 64 * sizeof(f16);
  f16* kf     = (f16*)ws;  ws += (size_t)32 * 2048 * 64 * sizeof(f16);
  f16* vf     = (f16*)ws;  ws += (size_t)32 * 2048 * 64 * sizeof(f16);
  f16* attnf  = (f16*)ws;

  const int gemmLds = 2 * 128 * 48 * (int)sizeof(f16);  // 24576 B

  cvt_f32_f16<<<4194304 / 256, 256, 0, stream>>>(x, xf, 4194304);
  cvt_f32_f16<<<786432 / 256, 256, 0, stream>>>(W_qkv, wqkvf, 786432);
  cvt_f32_f16<<<262144 / 256, 256, 0, stream>>>(W_out, woutf, 262144);

  // qkv = x @ W_qkv : M=8192, N=1536, K=512 (128x128 tiles)
  gemm_f16f32<<<dim3(1536 / 128, 8192 / 128), 256, gemmLds, stream>>>(
      xf, wqkvf, qkv, 1536, 512);

  rotary_split<<<2097152 / 256, 256, 0, stream>>>(qkv, qf, kf, vf);

  flash_attn<<<dim3(2048 / 64, 32), 256, FA_SMEM, stream>>>(qf, kf, vf,
                                                            pos_bias, attnf);

  // out = attn @ W_out : M=8192, N=512, K=512
  gemm_f16f32<<<dim3(512 / 128, 8192 / 128), 256, gemmLds, stream>>>(
      attnf, woutf, out, 512, 512);
}
